// MultiHeadAttention_21277267984993
// MI455X (gfx1250) — matrix operand order repaired and checked by reading, not compiled, not run on hardware
//
#include <hip/hip_runtime.h>

typedef __attribute__((ext_vector_type(16))) _Float16 v16h;
typedef __attribute__((ext_vector_type(8)))  _Float16 v8h;
typedef __attribute__((ext_vector_type(8)))  float    v8f;

#define EMB   768
#define HEADS 8
#define BATCH 8
#define SEQ   1024
#define DHEAD 96
#define BN    (BATCH * SEQ)   // 8192 rows

// ---------------------------------------------------------------------------
// WMMA helpers (gfx1250, wave32). Layouts used here (A/B k order contiguous; C/D per CDNA5 ISA):
//   A (16x32 f16): lane l -> row l%16; elem j -> K = j + 16*(l>=16)
//                  = one contiguous 16-half run at [hi*16]
//   B (32x16 f16): lane l -> col l%16; elem j -> K = j + 16*(l>=16)
//                  = one contiguous 16-half run at [hi*16] (when K-major per col)
//   C/D (16x16 f32): lane l -> col l%16; elem j -> row M = j + 8*(l>=16)
// ---------------------------------------------------------------------------
__device__ __forceinline__ v8f wmma_f16(v16h a, v16h b, v8f c) {
  return __builtin_amdgcn_wmma_f32_16x16x32_f16(
      false, a, false, b, (short)0, c, false, false);
}

// A-fragment: base points at the 32-half K-row for this lane's M row
__device__ __forceinline__ v16h frag_a(const _Float16* base, int hi) {
  v8h lo = *(const v8h*)(base + hi * 16);
  v8h hv = *(const v8h*)(base + hi * 16 + 8);
  return __builtin_shufflevector(lo, hv, 0, 1, 2, 3, 4, 5, 6, 7,
                                 8, 9, 10, 11, 12, 13, 14, 15);
}
// B-fragment: base points at the 32-half K-run for this lane's N column
__device__ __forceinline__ v16h frag_b(const _Float16* base, int hi) {
  v8h lo = *(const v8h*)(base + hi * 16);
  v8h hv = *(const v8h*)(base + hi * 16 + 8);
  return __builtin_shufflevector(lo, hv, 0, 1, 2, 3, 4, 5, 6, 7,
                                 8, 9, 10, 11, 12, 13, 14, 15);
}

// CDNA5 async global->LDS copy (bypasses VGPRs, tracked by ASYNCcnt)
__device__ __forceinline__ void async_b128(const _Float16* lds_dst,
                                           const _Float16* gsrc) {
  unsigned int la = (unsigned int)(unsigned long long)lds_dst;   // low 32b = LDS offset
  unsigned long long ga = (unsigned long long)gsrc;
  asm volatile("global_load_async_to_lds_b128 %0, %1, off"
               :: "v"(la), "v"(ga) : "memory");
}
__device__ __forceinline__ void wait_async() {
  asm volatile("s_wait_asynccnt 0" ::: "memory");
}

__device__ __forceinline__ float rmax16(float v) {
  v = fmaxf(v, __shfl_xor(v, 1, 16));
  v = fmaxf(v, __shfl_xor(v, 2, 16));
  v = fmaxf(v, __shfl_xor(v, 4, 16));
  v = fmaxf(v, __shfl_xor(v, 8, 16));
  return v;
}
__device__ __forceinline__ float rsum16(float v) {
  v += __shfl_xor(v, 1, 16);
  v += __shfl_xor(v, 2, 16);
  v += __shfl_xor(v, 4, 16);
  v += __shfl_xor(v, 8, 16);
  return v;
}

// ---------------------------------------------------------------------------
// Elementwise f32 -> f16 convert (used for x and all four W; y = x @ W.T means
// row-major W *is* the N-major B operand -> no transpose pass needed)
// ---------------------------------------------------------------------------
__global__ void k_cvt_f16(const float* __restrict__ s, _Float16* __restrict__ d, int n) {
  int i = blockIdx.x * blockDim.x + threadIdx.x;
  if (i < n) d[i] = (_Float16)s[i];
}

// ---------------------------------------------------------------------------
// Tiled WMMA GEMM: C[BN x EMB] = A[BN x EMB](f16) @ W^T + bias,
// with W supplied row-major f16 (N-major B). Block tile 32(M) x 128(N),
// 8 waves (2x4), each wave does a 16x32 slab = 2 WMMAs per K-step of 32.
// LDS: sA [32m][32k], sB [128n][32k]; both filled by async b128 copies.
// ---------------------------------------------------------------------------
template <bool HEAD_SCATTER>
__global__ __launch_bounds__(256) void k_gemm_wmma(
    const _Float16* __restrict__ A, const _Float16* __restrict__ W16,
    const float* __restrict__ bias, void* __restrict__ outp) {
  __shared__ __align__(16) _Float16 sA[32 * 32];
  __shared__ __align__(16) _Float16 sB[128 * 32];

  const int tid  = threadIdx.x;
  const int lane = tid & 31, l16 = lane & 15, hi = lane >> 4;
  const int wave = tid >> 5;
  const int wm   = wave >> 2;        // 0..1
  const int wn   = wave & 3;         // 0..3
  const int rowBase = blockIdx.x * 32;
  const int colBase = blockIdx.y * 128;

  const v8f Z8 = {0.f, 0.f, 0.f, 0.f, 0.f, 0.f, 0.f, 0.f};
  v8f acc0 = Z8, acc1 = Z8;

  for (int kk = 0; kk < EMB; kk += 32) {
    __syncthreads();
    // A tile: 32 rows x 32 halves = 128 b128 chunks
    for (int i = tid; i < 128; i += 256) {
      int r = i >> 2, c8 = (i & 3) << 3;
      async_b128(&sA[i * 8], A + (size_t)(rowBase + r) * EMB + kk + c8);
    }
    // B tile (n-major): 128 rows x 32 halves = 512 b128 chunks
    for (int i = tid; i < 512; i += 256) {
      int nr = i >> 2, c8 = (i & 3) << 3;
      async_b128(&sB[i * 8], W16 + (size_t)(colBase + nr) * EMB + kk + c8);
    }
    wait_async();
    __syncthreads();

    v16h a  = frag_a(&sA[(wm * 16 + l16) * 32], hi);
    v16h b0 = frag_b(&sB[(wn * 32 + l16) * 32], hi);
    v16h b1 = frag_b(&sB[(wn * 32 + 16 + l16) * 32], hi);
    acc0 = wmma_f16(a, b0, acc0);
    acc1 = wmma_f16(a, b1, acc1);
  }

#pragma unroll
  for (int t = 0; t < 2; ++t) {
    v8f acc = t ? acc1 : acc0;
#pragma unroll
    for (int j = 0; j < 8; ++j) {
      int rg = rowBase + wm * 16 + j + 8 * hi;       // global row in [0, BN)
      int cg = colBase + wn * 32 + t * 16 + l16;     // global col in [0, EMB)
      float val = acc[j] + bias[cg];
      if constexpr (HEAD_SCATTER) {
        // replicate reference's raw (b,n,hd)->(b,h,n,d) reflatten
        int bb = rg >> 10, rn = rg & 1023;
        int hh = rn >> 7, rr = rn & 127;
        int s = cg / DHEAD, di = cg % DHEAD;
        int ni = rr * 8 + s;
        ((_Float16*)outp)[(((size_t)bb * HEADS + hh) * SEQ + ni) * DHEAD + di] =
            (_Float16)val;
      } else {
        ((float*)outp)[(size_t)rg * EMB + cg] = val;
      }
    }
  }
}

// ---------------------------------------------------------------------------
// Flash attention per (b,h): softmax(Q K^T) / sqrt(EMB) @ V
// 4 waves/block, each owns a 16-row Q tile; key blocks of 32 async-staged in
// LDS. Output scattered into matrix layout Om[BN x EMB] f16 (inverse reshape).
// ---------------------------------------------------------------------------
__global__ __launch_bounds__(128) void k_flash_attn(
    const _Float16* __restrict__ Qh, const _Float16* __restrict__ Kh,
    const _Float16* __restrict__ Vh, _Float16* __restrict__ Om) {
  __shared__ __align__(16) _Float16 sK[32 * DHEAD];
  __shared__ __align__(16) _Float16 sV[32 * DHEAD];
  __shared__ __align__(16) _Float16 sP[4][16 * 32];

  const int tid  = threadIdx.x;
  const int lane = tid & 31, l16 = lane & 15, hi = lane >> 4;
  const int w    = tid >> 5;
  const int bx   = blockIdx.x;
  const int qt   = bx & 15;        // 64-row q-tile group
  const int bh   = bx >> 4;        // b*HEADS + h
  const int bb   = bh >> 3, hh = bh & 7;
  const size_t headOff = (size_t)bh * SEQ * DHEAD;
  const int qbase = qt * 64 + w * 16;

  // Q fragments: 16 rows x 96 feats = 3 A-frags (vectorized 16B runs)
  v16h aq[3];
  {
    const _Float16* qrow = Qh + headOff + (size_t)(qbase + l16) * DHEAD;
#pragma unroll
    for (int ks = 0; ks < 3; ++ks) aq[ks] = frag_a(qrow + ks * 32, hi);
  }

  float mrow[8], lsum[8];
  v8f o[6];
  const v8f Z8 = {0.f, 0.f, 0.f, 0.f, 0.f, 0.f, 0.f, 0.f};
#pragma unroll
  for (int j = 0; j < 8; ++j) { mrow[j] = -1e30f; lsum[j] = 0.f; }
#pragma unroll
  for (int f = 0; f < 6; ++f) o[f] = Z8;

  _Float16* sPw = &sP[w][0];

  for (int kb = 0; kb < SEQ / 32; ++kb) {
    __syncthreads();
    const _Float16* Kb = Kh + headOff + (size_t)kb * 32 * DHEAD;
    const _Float16* Vb = Vh + headOff + (size_t)kb * 32 * DHEAD;
    // 32 keys x 96 dims contiguous: 384 b128 chunks each, async to LDS
    for (int i = tid; i < 384; i += 128) {
      async_b128(&sK[i * 8], Kb + i * 8);
      async_b128(&sV[i * 8], Vb + i * 8);
    }
    if (kb + 1 < SEQ / 32)  // emits global_prefetch_b8
      __builtin_prefetch(Kb + 32 * DHEAD, 0, 1);
    wait_async();
    __syncthreads();

    // S = Q(16x96) @ Kblk^T(96x32): key-major sK rows give contiguous B-frags
    v8f s0 = Z8, s1 = Z8;
#pragma unroll
    for (int ks = 0; ks < 3; ++ks) {
      v16h bk0 = frag_b(&sK[l16 * DHEAD + ks * 32], hi);
      v16h bk1 = frag_b(&sK[(16 + l16) * DHEAD + ks * 32], hi);
      s0 = wmma_f16(aq[ks], bk0, s0);
      s1 = wmma_f16(aq[ks], bk1, s1);
    }

    // Online softmax; rows live in C-layout (row = j+8*hi, cols across 16 lanes)
#pragma unroll
    for (int j = 0; j < 8; ++j) {
      float v = rmax16(fmaxf(s0[j], s1[j]));
      float mnew = fmaxf(mrow[j], v);
      float scale = __expf(mrow[j] - mnew);
      mrow[j] = mnew;
      lsum[j] *= scale;
#pragma unroll
      for (int f = 0; f < 6; ++f) o[f][j] *= scale;
      float p0 = __expf(s0[j] - mnew);
      float p1 = __expf(s1[j] - mnew);
      lsum[j] += rsum16(p0 + p1);
      // stage P (16 rows x 32 keys, row-major) wave-privately: C->A relayout
      sPw[(j + 8 * hi) * 32 + l16]      = (_Float16)p0;
      sPw[(j + 8 * hi) * 32 + 16 + l16] = (_Float16)p1;
    }

    v16h ap = frag_a(&sPw[l16 * 32], hi);   // same-wave LDS: in-order, no barrier

    // O(16x96) += P(16x32) @ Vblk(32x96)
#pragma unroll
    for (int f = 0; f < 6; ++f) {
      v16h bv;
#pragma unroll
      for (int j = 0; j < 16; ++j)
        bv[j] = sV[(j + hi * 16) * DHEAD + f * 16 + l16];
      o[f] = wmma_f16(ap, bv, o[f]);
    }
  }

  // normalize by row sum, apply post-softmax /sqrt(EMB), scatter into matrix
  // layout: row = b*SEQ + h*128 + n/8, col = (n%8)*96 + d
  const float sc = 27.712812921102035f;  // sqrt(768)
#pragma unroll
  for (int j = 0; j < 8; ++j) {
    float inv = 1.0f / (lsum[j] * sc);
    int ni = qbase + j + 8 * hi;
    int r = bb * SEQ + hh * 128 + (ni >> 3);
    int cb = (ni & 7) * DHEAD;
#pragma unroll
    for (int f = 0; f < 6; ++f)
      Om[(size_t)r * EMB + cb + f * 16 + l16] = (_Float16)(o[f][j] * inv);
  }
}

// ---------------------------------------------------------------------------
// Host-side orchestration (graph-capture safe: kernels only, all on `stream`)
// ---------------------------------------------------------------------------
extern "C" void kernel_launch(void* const* d_in, const int* in_sizes, int n_in,
                              void* d_out, int out_size, void* d_ws, size_t ws_size,
                              hipStream_t stream) {
  const float* x  = (const float*)d_in[0];
  const float* Wq = (const float*)d_in[1];
  const float* bq = (const float*)d_in[2];
  const float* Wk = (const float*)d_in[3];
  const float* bk = (const float*)d_in[4];
  const float* Wv = (const float*)d_in[5];
  const float* bv = (const float*)d_in[6];
  const float* Wp = (const float*)d_in[7];
  const float* bp = (const float*)d_in[8];

  char* p = (char*)d_ws;
  const size_t SZ_XH = (size_t)BN * EMB * sizeof(_Float16);   // 12.6 MB
  const size_t SZ_W  = (size_t)EMB * EMB * sizeof(_Float16);  // 1.18 MB

  _Float16* xh   = (_Float16*)p;
  _Float16* wq16 = (_Float16*)(p + SZ_XH);
  _Float16* wk16 = wq16 + (size_t)EMB * EMB;
  _Float16* wv16 = wk16 + (size_t)EMB * EMB;
  _Float16* wp16 = wv16 + (size_t)EMB * EMB;
  _Float16* Qh   = (_Float16*)(p + SZ_XH + 4 * SZ_W);
  _Float16* Kh   = Qh + (size_t)BN * EMB;
  _Float16* Vh   = Kh + (size_t)BN * EMB;
  _Float16* Om   = Vh + (size_t)BN * EMB;
  // total workspace ~68 MB

  const int nx = BN * EMB;
  k_cvt_f16<<<(nx + 255) / 256, 256, 0, stream>>>(x, xh, nx);
  const int nw = EMB * EMB;
  k_cvt_f16<<<(nw + 255) / 256, 256, 0, stream>>>(Wq, wq16, nw);
  k_cvt_f16<<<(nw + 255) / 256, 256, 0, stream>>>(Wk, wk16, nw);
  k_cvt_f16<<<(nw + 255) / 256, 256, 0, stream>>>(Wv, wv16, nw);
  k_cvt_f16<<<(nw + 255) / 256, 256, 0, stream>>>(Wp, wp16, nw);

  dim3 gg(BN / 32, EMB / 128);  // 256 x 6 blocks
  k_gemm_wmma<true><<<gg, 256, 0, stream>>>(xh, wq16, bq, (void*)Qh);
  k_gemm_wmma<true><<<gg, 256, 0, stream>>>(xh, wk16, bk, (void*)Kh);
  k_gemm_wmma<true><<<gg, 256, 0, stream>>>(xh, wv16, bv, (void*)Vh);

  k_flash_attn<<<BATCH * HEADS * (SEQ / 64), 128, 0, stream>>>(Qh, Kh, Vh, Om);

  k_gemm_wmma<false><<<gg, 256, 0, stream>>>(Om, wp16, bp, d_out);
}
